// QLDGCNN_semseg_s3dis_61890478735421
// MI455X (gfx1250) — matrix operand order repaired and checked by reading, not compiled, not run on hardware
//
#include <hip/hip_runtime.h>
#include <hip/hip_bf16.h>
#include <math.h>

typedef __bf16 bf16_t;
typedef __bf16 v8bf  __attribute__((ext_vector_type(8)));
typedef __bf16 v16bf __attribute__((ext_vector_type(16)));
typedef float  v8f   __attribute__((ext_vector_type(8)));

#define TILE_P 64
#define TILE_O 64
#define KSTRIDE 40   // LDS row stride in bf16 elements: 80B, keeps 16B alignment
#define EPSB 1e-5f

// Build a 16-element bf16 fragment from two 16B-aligned LDS chunks
// (each chunk lowers to a single ds_load_b128).
__device__ __forceinline__ v16bf load_frag16(const bf16_t* p0, const bf16_t* p1)
{
    v8bf lo = *(const v8bf*)p0;
    v8bf hi = *(const v8bf*)p1;
    v16bf r;
#pragma unroll
    for (int i = 0; i < 8; ++i) {
        r[i]     = lo[i];
        r[i + 8] = hi[i];
    }
    return r;
}

// ---------------------------------------------------------------------------
// Fused GEMM (+ optional quadratic combine, BN, LeakyReLU) using bf16 WMMA.
// MODE: 0 = edge features built on the fly from f32 cat + knn idx
//       1 = plain bf16 activation input
//       2 = plain f32 activation input (converted on load)
// QUAD: 1 -> y = (Wr x)*(Wg x + bg) + Wb x^2 ; 0 -> y = Wr x
// ---------------------------------------------------------------------------
template <int MODE, int QUAD, int DOBN>
__global__ __launch_bounds__(256)
void fused_gemm_kernel(
    const float* __restrict__ catX, int catBatchStride,
    const bf16_t* __restrict__ Xb,
    const float* __restrict__ Xf,
    int xBatchStride,
    const int* __restrict__ idx, int N, int Kn,
    int Cin, int P,
    const float* __restrict__ wr, const float* __restrict__ wg,
    const float* __restrict__ bg, const float* __restrict__ wb,
    const float* __restrict__ gamma, const float* __restrict__ beta,
    const float* __restrict__ mean, const float* __restrict__ var,
    int CoutReal,
    bf16_t* __restrict__ Yb, float* __restrict__ Yf, int yBatchStride)
{
    // K-contiguous tiles: X transposed to [p][k], weights [o][k].
    __shared__ __attribute__((aligned(16))) bf16_t sXt [TILE_P][KSTRIDE];
    __shared__ __attribute__((aligned(16))) bf16_t sXqt[QUAD ? TILE_P : 1][KSTRIDE];
    __shared__ __attribute__((aligned(16))) bf16_t sWr [TILE_O][KSTRIDE];
    __shared__ __attribute__((aligned(16))) bf16_t sWg [QUAD ? TILE_O : 1][KSTRIDE];
    __shared__ __attribute__((aligned(16))) bf16_t sWb [QUAD ? TILE_O : 1][KSTRIDE];

    const int b     = blockIdx.z;
    const int pBase = blockIdx.x * TILE_P;
    const int oBase = blockIdx.y * TILE_O;
    const int tid   = threadIdx.x;
    const int lane  = tid & 31;
    const int wid   = tid >> 5;
    const int mT    = wid & 3;   // 4 row blocks of 16
    const int nT    = wid >> 2;  // 2 col blocks of 32 (2 sub-tiles of 16 each)
    const int hi    = lane >> 4;
    const int ln    = lane & 15;

    const float*  cat  = (MODE == 0) ? catX + (size_t)b * catBatchStride : nullptr;
    const bf16_t* Xbb  = (MODE == 1) ? Xb + (size_t)b * xBatchStride : nullptr;
    const float*  Xff  = (MODE == 2) ? Xf + (size_t)b * xBatchStride : nullptr;
    const int*    idxb = (MODE == 0) ? idx + (size_t)b * N * Kn : nullptr;

    v8f accR[2] = {};
    v8f accG[2] = {};
    v8f accB[2] = {};

    const int Ce = Cin >> 1;  // half channels for edge mode

    for (int kb = 0; kb < Cin; kb += 32) {
        // ---- stage activation tile (and its square) into LDS, [p][k] -------
        for (int e = tid; e < 32 * TILE_P; e += 256) {
            int kl  = e >> 6;          // 0..31
            int pl  = e & 63;          // 0..63
            int cin = kb + kl;
            int p   = pBase + pl;
            float v = 0.f;
            if (cin < Cin && p < P) {
                if (MODE == 0) {
                    int n  = p / Kn;
                    int kk = p - n * Kn;
                    if (cin < Ce) {
                        int j = idxb[n * Kn + kk];
                        v = cat[(size_t)cin * N + j] - cat[(size_t)cin * N + n];
                    } else {
                        v = cat[(size_t)(cin - Ce) * N + n];
                    }
                } else if (MODE == 1) {
                    v = (float)Xbb[(size_t)cin * P + p];
                    if (kb + 32 < Cin)
                        __builtin_prefetch(&Xbb[(size_t)(cin + 32) * P + p], 0, 1);
                } else {
                    v = Xff[(size_t)cin * P + p];
                    if (kb + 32 < Cin)
                        __builtin_prefetch(&Xff[(size_t)(cin + 32) * P + p], 0, 1);
                }
            }
            sXt[pl][kl] = (bf16_t)v;
            if (QUAD) sXqt[pl][kl] = (bf16_t)(v * v);
        }
        // ---- stage weight tiles (f32 -> bf16), [o][k] ----------------------
        for (int e = tid; e < TILE_O * 32; e += 256) {
            int ol  = e >> 5;
            int kl  = e & 31;
            int o   = oBase + ol;
            int cin = kb + kl;
            float vr = 0.f, vg = 0.f, vb = 0.f;
            if (o < CoutReal && cin < Cin) {
                vr = wr[(size_t)o * Cin + cin];
                if (QUAD) {
                    vg = wg[(size_t)o * Cin + cin];
                    vb = wb[(size_t)o * Cin + cin];
                }
                if (kb + 32 < Cin)
                    __builtin_prefetch(&wr[(size_t)o * Cin + cin + 32], 0, 1);
            }
            sWr[ol][kl] = (bf16_t)vr;
            if (QUAD) {
                sWg[ol][kl] = (bf16_t)vg;
                sWb[ol][kl] = (bf16_t)vb;
            }
        }
        __syncthreads();

        // ---- fragments: two ds_load_b128 chunks each ------------------------
        const int m  = mT * 16 + ln;
        const int n0 = nT * 32 + ln;        // first 16-col sub-tile
        const int n1 = n0 + 16;             // second 16-col sub-tile

        v16bf aR  = load_frag16(&sWr[m][16 * hi], &sWr[m][16 * hi + 8]);
        v16bf bX0 = load_frag16(&sXt[n0][16 * hi], &sXt[n0][16 * hi + 8]);
        v16bf bX1 = load_frag16(&sXt[n1][16 * hi], &sXt[n1][16 * hi + 8]);

        accR[0] = __builtin_amdgcn_wmma_f32_16x16x32_bf16(
            false, aR, false, bX0, (short)0, accR[0], false, false);
        accR[1] = __builtin_amdgcn_wmma_f32_16x16x32_bf16(
            false, aR, false, bX1, (short)0, accR[1], false, false);

        if (QUAD) {
            v16bf aG  = load_frag16(&sWg[m][16 * hi], &sWg[m][16 * hi + 8]);
            v16bf aB  = load_frag16(&sWb[m][16 * hi], &sWb[m][16 * hi + 8]);
            v16bf bQ0 = load_frag16(&sXqt[n0][16 * hi], &sXqt[n0][16 * hi + 8]);
            v16bf bQ1 = load_frag16(&sXqt[n1][16 * hi], &sXqt[n1][16 * hi + 8]);

            accG[0] = __builtin_amdgcn_wmma_f32_16x16x32_bf16(
                false, aG, false, bX0, (short)0, accG[0], false, false);
            accG[1] = __builtin_amdgcn_wmma_f32_16x16x32_bf16(
                false, aG, false, bX1, (short)0, accG[1], false, false);
            accB[0] = __builtin_amdgcn_wmma_f32_16x16x32_bf16(
                false, aB, false, bQ0, (short)0, accB[0], false, false);
            accB[1] = __builtin_amdgcn_wmma_f32_16x16x32_bf16(
                false, aB, false, bQ1, (short)0, accB[1], false, false);
        }
        __syncthreads();
    }

    // ---- epilogue: quadratic combine + BN + LeakyReLU ----------------------
#pragma unroll
    for (int t = 0; t < 2; ++t) {
#pragma unroll
        for (int r = 0; r < 8; ++r) {
            int o = oBase + mT * 16 + r + 8 * hi;
            int p = pBase + nT * 32 + t * 16 + ln;
            if (o < CoutReal && p < P) {
                float y;
                if (QUAD) {
                    float gv = accG[t][r] + bg[o];
                    y = accR[t][r] * gv + accB[t][r];
                } else {
                    y = accR[t][r];
                }
                if (DOBN) {
                    y = (y - mean[o]) * rsqrtf(var[o] + EPSB);
                    y = y * gamma[o] + beta[o];
                    y = y > 0.f ? y : 0.2f * y;
                }
                if (Yb) Yb[(size_t)b * yBatchStride + (size_t)o * P + p] = (bf16_t)y;
                if (Yf) Yf[(size_t)b * yBatchStride + (size_t)o * P + p] = y;
            }
        }
    }
}

// ---------------------------------------------------------------------------
// Fused KNN: per-point distance row in LDS + 20 argmin rounds (no NxN in HBM)
// ---------------------------------------------------------------------------
#define KNN_THREADS 256
__global__ __launch_bounds__(KNN_THREADS)
void knn_kernel(const float* __restrict__ cat, int catBatchStride,
                int rowOff, int C, int N, int Kn, int* __restrict__ idxOut)
{
    __shared__ float q[160];
    __shared__ float dist[4096];
    __shared__ float rbest[KNN_THREADS];
    __shared__ int   rarg[KNN_THREADS];

    const int gid = blockIdx.x;
    const int b = gid / N;
    const int n = gid - b * N;
    const float* base = cat + (size_t)b * catBatchStride + (size_t)rowOff * N;
    const int tid = threadIdx.x;

    for (int c = tid; c < C; c += KNN_THREADS) q[c] = base[(size_t)c * N + n];
    __syncthreads();

    for (int m = tid; m < N; m += KNN_THREADS) {
        float s = 0.f;
        for (int c = 0; c < C; ++c) {
            float d = base[(size_t)c * N + m] - q[c];
            s += d * d;
        }
        dist[m] = s;
    }
    __syncthreads();

    int* out = idxOut + (size_t)(b * N + n) * Kn;
    for (int t = 0; t < Kn; ++t) {
        float best = 3.4e38f;
        int barg = 0;
        for (int m = tid; m < N; m += KNN_THREADS) {
            float d = dist[m];
            if (d < best) { best = d; barg = m; }
        }
        rbest[tid] = best;
        rarg[tid] = barg;
        __syncthreads();
        for (int s = KNN_THREADS / 2; s > 0; s >>= 1) {
            if (tid < s && rbest[tid + s] < rbest[tid]) {
                rbest[tid] = rbest[tid + s];
                rarg[tid]  = rarg[tid + s];
            }
            __syncthreads();
        }
        if (tid == 0) {
            out[t] = rarg[0];
            dist[rarg[0]] = 3.4e38f;
        }
        __syncthreads();
    }
}

// ---------------------------------------------------------------------------
// Small helper kernels
// ---------------------------------------------------------------------------
__global__ void copy_x_kernel(const float* __restrict__ x,
                              float* __restrict__ cat201, int N)
{
    int i = blockIdx.x * blockDim.x + threadIdx.x;
    int total = 4 * 9 * N;
    if (i < total) {
        int b = i / (9 * N);
        int rem = i - b * 9 * N;
        cat201[(size_t)b * 201 * N + rem] = x[i];
    }
}

__global__ void maxpool_k_kernel(const bf16_t* __restrict__ Y,
                                 float* __restrict__ cat201,
                                 int rowOff, int N, int Kn)
{
    int i = blockIdx.x * blockDim.x + threadIdx.x;
    int total = 4 * 64 * N;
    if (i < total) {
        int b = i / (64 * N);
        int rem = i - b * 64 * N;
        int c = rem / N;
        int n = rem - c * N;
        const bf16_t* src = Y + ((size_t)b * 64 + c) * (size_t)(N * Kn) + (size_t)n * Kn;
        float mx = -3.4e38f;
        for (int k = 0; k < Kn; ++k) {
            float v = (float)src[k];
            mx = v > mx ? v : mx;
        }
        cat201[((size_t)b * 201 + rowOff + c) * N + n] = mx;
    }
}

__global__ void rowmax_kernel(const bf16_t* __restrict__ h9,
                              float* __restrict__ g, int N)
{
    int i = blockIdx.x * blockDim.x + threadIdx.x;
    if (i < 4 * 1024) {
        const bf16_t* src = h9 + (size_t)i * N;
        float mx = -3.4e38f;
        for (int n = 0; n < N; ++n) {
            float v = (float)src[n];
            mx = v > mx ? v : mx;
        }
        g[i] = mx;
    }
}

__global__ void build_cat1216_kernel(const float* __restrict__ g,
                                     const float* __restrict__ cat201,
                                     bf16_t* __restrict__ out, int N)
{
    int i = blockIdx.x * blockDim.x + threadIdx.x;
    int total = 4 * 1216 * N;
    if (i < total) {
        int b = i / (1216 * N);
        int rem = i - b * 1216 * N;
        int ch = rem / N;
        int n = rem - ch * N;
        float v = (ch < 1024)
                      ? g[b * 1024 + ch]
                      : cat201[((size_t)b * 201 + 9 + (ch - 1024)) * N + n];
        out[i] = (bf16_t)v;
    }
}

// ---------------------------------------------------------------------------
// Host side
// ---------------------------------------------------------------------------
static void launch_gemm(hipStream_t s, int mode, int quadMode, int doBn,
                        const float* cat, int catBS,
                        const bf16_t* Xb, const float* Xf, int xBS,
                        const int* idx, int N, int Kn, int Cin, int P,
                        const float* wr, const float* wg, const float* bg,
                        const float* wb, const float* gamma, const float* beta,
                        const float* mean, const float* var,
                        int Cout, int CoutReal,
                        bf16_t* Yb, float* Yf, int yBS)
{
    dim3 grid((P + TILE_P - 1) / TILE_P, Cout / TILE_O, 4);
#define ARGS cat, catBS, Xb, Xf, xBS, idx, N, Kn, Cin, P, \
             wr, wg, bg, wb, gamma, beta, mean, var, CoutReal, Yb, Yf, yBS
    if (mode == 0 && quadMode == 1 && doBn == 1)
        fused_gemm_kernel<0, 1, 1><<<grid, 256, 0, s>>>(ARGS);
    else if (mode == 2 && quadMode == 1 && doBn == 1)
        fused_gemm_kernel<2, 1, 1><<<grid, 256, 0, s>>>(ARGS);
    else if (mode == 1 && quadMode == 1 && doBn == 1)
        fused_gemm_kernel<1, 1, 1><<<grid, 256, 0, s>>>(ARGS);
    else if (mode == 1 && quadMode == 0 && doBn == 1)
        fused_gemm_kernel<1, 0, 1><<<grid, 256, 0, s>>>(ARGS);
    else
        fused_gemm_kernel<1, 0, 0><<<grid, 256, 0, s>>>(ARGS);
#undef ARGS
}

extern "C" void kernel_launch(void* const* d_in, const int* in_sizes, int n_in,
                              void* d_out, int out_size, void* d_ws, size_t ws_size,
                              hipStream_t stream)
{
    (void)in_sizes; (void)n_in; (void)out_size; (void)ws_size;

    const int B = 4, N = 4096, KN = 20;
    const int P2 = N * KN;  // 81920 edge positions

    const float* x = (const float*)d_in[0];
    // quad param blocks: wr,wg,bg,wb,gamma,beta,mean,var
    // c1=1 c2=9 c3=17 c4=25 c5=33 c6=41 c9=49 ; c10=57(w,g,b,m,v) c11=62 ; w12=67
    auto F = [&](int i) { return (const float*)d_in[i]; };

    // ---- workspace layout --------------------------------------------------
    char* ws = (char*)d_ws;
    size_t off = 0;
    bf16_t* h1      = (bf16_t*)(ws + off); off += (size_t)B * 64 * P2 * 2;
    bf16_t* h2      = (bf16_t*)(ws + off); off += (size_t)B * 64 * P2 * 2;
    float*  cat201  = (float*)(ws + off);  off += (size_t)B * 201 * N * 4;
    int*    idx     = (int*)(ws + off);    off += (size_t)B * N * KN * 4;
    bf16_t* h9      = (bf16_t*)(ws + off); off += (size_t)B * 1024 * N * 2;
    bf16_t* cat1216 = (bf16_t*)(ws + off); off += (size_t)B * 1216 * N * 2;
    float*  g       = (float*)(ws + off);  off += (size_t)B * 1024 * 4;
    bf16_t* h10 = h1;   // reuse: stage activations are dead by now
    bf16_t* h11 = h2;

    const int catBS = 201 * N;

    // ---- seed cat201 rows 0..8 with x --------------------------------------
    {
        int total = B * 9 * N;
        copy_x_kernel<<<(total + 255) / 256, 256, 0, stream>>>(x, cat201, N);
    }

    // ---- stage 1: knn on rows 6..8, edge conv c1/c2, maxpool -> rows 9..72 -
    knn_kernel<<<B * N, KNN_THREADS, 0, stream>>>(cat201, catBS, 6, 3, N, KN, idx);
    launch_gemm(stream, 0, 1, 1, cat201, catBS, nullptr, nullptr, 0, idx, N, KN,
                18, P2, F(1), F(2), F(3), F(4), F(5), F(6), F(7), F(8),
                64, 64, h1, nullptr, 64 * P2);
    launch_gemm(stream, 1, 1, 1, nullptr, 0, h1, nullptr, 64 * P2, nullptr, N, KN,
                64, P2, F(9), F(10), F(11), F(12), F(13), F(14), F(15), F(16),
                64, 64, h2, nullptr, 64 * P2);
    maxpool_k_kernel<<<(B * 64 * N + 255) / 256, 256, 0, stream>>>(h2, cat201, 9, N, KN);

    // ---- stage 2: knn on rows 0..72 (73ch), c3/c4, maxpool -> rows 73..136 -
    knn_kernel<<<B * N, KNN_THREADS, 0, stream>>>(cat201, catBS, 0, 73, N, KN, idx);
    launch_gemm(stream, 0, 1, 1, cat201, catBS, nullptr, nullptr, 0, idx, N, KN,
                146, P2, F(17), F(18), F(19), F(20), F(21), F(22), F(23), F(24),
                64, 64, h1, nullptr, 64 * P2);
    launch_gemm(stream, 1, 1, 1, nullptr, 0, h1, nullptr, 64 * P2, nullptr, N, KN,
                64, P2, F(25), F(26), F(27), F(28), F(29), F(30), F(31), F(32),
                64, 64, h2, nullptr, 64 * P2);
    maxpool_k_kernel<<<(B * 64 * N + 255) / 256, 256, 0, stream>>>(h2, cat201, 73, N, KN);

    // ---- stage 3: knn on rows 0..136 (137ch), c5/c6, maxpool -> rows 137..200
    knn_kernel<<<B * N, KNN_THREADS, 0, stream>>>(cat201, catBS, 0, 137, N, KN, idx);
    launch_gemm(stream, 0, 1, 1, cat201, catBS, nullptr, nullptr, 0, idx, N, KN,
                274, P2, F(33), F(34), F(35), F(36), F(37), F(38), F(39), F(40),
                64, 64, h1, nullptr, 64 * P2);
    launch_gemm(stream, 1, 1, 1, nullptr, 0, h1, nullptr, 64 * P2, nullptr, N, KN,
                64, P2, F(41), F(42), F(43), F(44), F(45), F(46), F(47), F(48),
                64, 64, h2, nullptr, 64 * P2);
    maxpool_k_kernel<<<(B * 64 * N + 255) / 256, 256, 0, stream>>>(h2, cat201, 137, N, KN);

    // ---- c9: quad1d 201 -> 1024 (f32 input read directly from cat201) ------
    launch_gemm(stream, 2, 1, 1, nullptr, 0, nullptr, cat201, catBS, nullptr, N, KN,
                201, N, F(49), F(50), F(51), F(52), F(53), F(54), F(55), F(56),
                1024, 1024, h9, nullptr, 1024 * N);

    // ---- global max + build [g | x1 | x2 | x3] (1216 ch) -------------------
    rowmax_kernel<<<(B * 1024 + 255) / 256, 256, 0, stream>>>(h9, g, N);
    build_cat1216_kernel<<<(B * 1216 * N + 255) / 256, 256, 0, stream>>>(g, cat201, cat1216, N);

    // ---- c10: 1216 -> 512 (conv+bn+lrelu) ----------------------------------
    launch_gemm(stream, 1, 0, 1, nullptr, 0, cat1216, nullptr, 1216 * N, nullptr, N, KN,
                1216, N, F(57), nullptr, nullptr, nullptr, F(58), F(59), F(60), F(61),
                512, 512, h10, nullptr, 512 * N);

    // ---- c11: 512 -> 256 ----------------------------------------------------
    launch_gemm(stream, 1, 0, 1, nullptr, 0, h10, nullptr, 512 * N, nullptr, N, KN,
                512, N, F(62), nullptr, nullptr, nullptr, F(63), F(64), F(65), F(66),
                256, 256, h11, nullptr, 256 * N);

    // ---- w12: 256 -> 13 (plain linear, f32 output to d_out) ----------------
    launch_gemm(stream, 1, 0, 0, nullptr, 0, h11, nullptr, 256 * N, nullptr, N, KN,
                256, N, F(67), nullptr, nullptr, nullptr, nullptr, nullptr, nullptr, nullptr,
                64, 13, nullptr, (float*)d_out, 13 * N);
}
